// SelfAttention_41472204210243
// MI455X (gfx1250) — compile-verified, matrix operand order checked by reading, not run on hardware
//
#include <hip/hip_runtime.h>
#include <hip/hip_bf16.h>

// ---------------- types ----------------
typedef __attribute__((ext_vector_type(16))) __bf16        v16bf;
typedef __attribute__((ext_vector_type(8)))  float         v8f;
typedef __attribute__((ext_vector_type(8)))  unsigned int  v8u;
typedef __attribute__((ext_vector_type(4)))  int           v4i;

#define LDSK 40            // padded K-stride of LDS tiles (32 + 8) -> 80B rows, 16B-aligned chunks
#define BM 128
#define BN 128
#define BK 32

#define AS1 __attribute__((address_space(1)))
#define AS3 __attribute__((address_space(3)))

// CDNA5 async global->LDS copy path (guarded; falls back to vmem+ds if absent)
#if defined(__has_builtin)
#if __has_builtin(__builtin_amdgcn_global_load_async_to_lds_b128)
#define HAVE_ASYNC 1
#endif
#endif
#ifndef HAVE_ASYNC
#define HAVE_ASYNC 0
#endif

static_assert(sizeof(__bf16) == 2, "bf16 size");

// fp32 -> bf16 round-to-nearest-even
__device__ __forceinline__ unsigned short f2bf(float f) {
  unsigned int u = __builtin_bit_cast(unsigned int, f);
  u = (u + 0x7FFFu + ((u >> 16) & 1u)) >> 16;
  return (unsigned short)u;
}

template <int N>
__device__ __forceinline__ void wait_async() {
#if HAVE_ASYNC
#if __has_builtin(__builtin_amdgcn_s_wait_asynccnt)
  __builtin_amdgcn_s_wait_asynccnt(N);
#else
  asm volatile("s_wait_asynccnt %0" ::"n"(N) : "memory");
#endif
#endif
}

// ---------------- WMMA fragment load (A row / B col), ISA 16-bit 16x32 layout ----------------
// lane L (0..31): row/col = L&15 ; K-base = (L>>4)*8 ; elems 0..7 -> K kb..kb+7, 8..15 -> 16+kb..
__device__ __forceinline__ v16bf load_frag(const unsigned short* tile, int lane) {
  const int r  = lane & 15;
  const int kb = (lane >> 4) * 8;
  const unsigned int* p0 = (const unsigned int*)(tile + r * LDSK + kb);
  const unsigned int* p1 = (const unsigned int*)(tile + r * LDSK + kb + 16);
  v8u raw;
  raw[0] = p0[0]; raw[1] = p0[1]; raw[2] = p0[2]; raw[3] = p0[3];
  raw[4] = p1[0]; raw[5] = p1[1]; raw[6] = p1[2]; raw[7] = p1[3];
  return __builtin_bit_cast(v16bf, raw);
}

// 8 waves: 2 (M) x 4 (N). Each wave: 64x32 = 4x2 grid of 16x16 WMMA tiles.
__device__ __forceinline__ void mma_step(const unsigned short* lds_a,
                                         const unsigned short* lds_b,
                                         v8f c[4][2], int wave, int lane) {
  const int wm = (wave >> 2) * 64;
  const int wn = (wave & 3) * 32;
  v16bf a[4], b[2];
#pragma unroll
  for (int i = 0; i < 4; ++i) a[i] = load_frag(lds_a + (wm + i * 16) * LDSK, lane);
#pragma unroll
  for (int j = 0; j < 2; ++j) b[j] = load_frag(lds_b + (wn + j * 16) * LDSK, lane);
#pragma unroll
  for (int i = 0; i < 4; ++i)
#pragma unroll
    for (int j = 0; j < 2; ++j)
      c[i][j] = __builtin_amdgcn_wmma_f32_16x16x32_bf16(
          false, a[i], false, b[j], (short)0, c[i][j], false, false);
}

// ---------------- tile loaders (global -> LDS) ----------------
// 128x32 f32 tile, row-major src, convert to bf16, LDS stride LDSK.
__device__ __forceinline__ void ld_f32_rows(unsigned short* lds, const float* src,
                                            size_t ld, int tid) {
#pragma unroll
  for (int q = 0; q < 4; ++q) {
    int f   = q * 256 + tid;        // float4 index, 1024 total
    int row = f >> 3;               // 8 float4 per 32-wide row
    int k   = (f & 7) * 4;
    float4 v = *(const float4*)(src + (size_t)row * ld + k);
    unsigned int lo = (unsigned int)f2bf(v.x) | ((unsigned int)f2bf(v.y) << 16);
    unsigned int hi = (unsigned int)f2bf(v.z) | ((unsigned int)f2bf(v.w) << 16);
    unsigned int* d = (unsigned int*)(lds + row * LDSK + k);
    d[0] = lo; d[1] = hi;
  }
}

// 32(K) x 128(N) f32 tile, transpose to LDS [n][k] as bf16.
__device__ __forceinline__ void ld_f32_trans(unsigned short* lds, const float* src,
                                             size_t ld, int tid) {
#pragma unroll
  for (int q = 0; q < 4; ++q) {
    int f  = q * 256 + tid;         // float4 index, 1024 total
    int kk = f >> 5;                // 32 float4 per 128-wide row
    int n  = (f & 31) * 4;
    float4 v = *(const float4*)(src + (size_t)kk * ld + n);
    lds[(n + 0) * LDSK + kk] = f2bf(v.x);
    lds[(n + 1) * LDSK + kk] = f2bf(v.y);
    lds[(n + 2) * LDSK + kk] = f2bf(v.z);
    lds[(n + 3) * LDSK + kk] = f2bf(v.w);
  }
}

// 128x32 bf16 tile, row-major src -> LDS rows (manual fallback path).
__device__ __forceinline__ void ld_bf16_rows(unsigned short* lds, const unsigned short* src,
                                             size_t ld, int tid) {
#pragma unroll
  for (int q = 0; q < 2; ++q) {
    int f   = q * 256 + tid;        // uint4 index (8 bf16), 512 total
    int row = f >> 2;
    int k   = (f & 3) * 8;
    uint4 v = *(const uint4*)(src + (size_t)row * ld + k);
    unsigned int* d = (unsigned int*)(lds + row * LDSK + k);
    d[0] = v.x; d[1] = v.y; d[2] = v.z; d[3] = v.w;
  }
}

// 128x32 bf16 tile: issue async DMA global->LDS (2 x b128 per lane per tile).
__device__ __forceinline__ void issue_tile(unsigned short* lds, const unsigned short* src,
                                           size_t ld, int tid) {
#if HAVE_ASYNC
#pragma unroll
  for (int q = 0; q < 2; ++q) {
    int f   = q * 256 + tid;
    int row = f >> 2;
    int k   = (f & 3) * 8;
    __builtin_amdgcn_global_load_async_to_lds_b128(
        (AS1 v4i*)(src + (size_t)row * ld + k),
        (AS3 v4i*)(lds + row * LDSK + k),
        0, 0);
  }
#else
  ld_bf16_rows(lds, src, ld, tid);
#endif
}

// ---------------- shared double-buffered bf16 GEMM core (A,B row-major [*][K]) ----------------
__device__ __forceinline__ void gemm_core(const unsigned short* A, size_t lda,
                                          const unsigned short* B, size_t ldb, int K,
                                          unsigned short lds_a[2][BM * LDSK],
                                          unsigned short lds_b[2][BN * LDSK],
                                          v8f c[4][2], int tid, int wave, int lane) {
  const int nit = K / BK;
  issue_tile(lds_a[0], A, lda, tid);
  issue_tile(lds_b[0], B, ldb, tid);
  for (int it = 0; it < nit; ++it) {
    const int cur = it & 1;
    if (it + 1 < nit) {
      issue_tile(lds_a[cur ^ 1], A + (size_t)(it + 1) * BK, lda, tid);
      issue_tile(lds_b[cur ^ 1], B + (size_t)(it + 1) * BK, ldb, tid);
      wait_async<4>();   // current buffer's 4 copies done; next 4 still in flight
    } else {
      wait_async<0>();
    }
    __syncthreads();
    mma_step(lds_a[cur], lds_b[cur], c, wave, lane);
    __syncthreads();
  }
}

// ---------------- sizes ----------------
#define BB 4
#define SS 2048
#define DD 1024
#define HH 1024
#define NROWS (BB * SS)   // 8192

// ============ kernel 1: q/k/v = x @ W + b, bf16 out (z selects; v stored transposed) ============
__global__ __launch_bounds__(256) void attn_qkv_kernel(
    const float* __restrict__ x,
    const float* __restrict__ Wq, const float* __restrict__ bq,
    const float* __restrict__ Wk, const float* __restrict__ bk,
    const float* __restrict__ Wv, const float* __restrict__ bv,
    unsigned short* __restrict__ qb, unsigned short* __restrict__ kb,
    unsigned short* __restrict__ vtb) {
  const float* W; const float* bias; unsigned short* out;
  if (blockIdx.z == 0)      { W = Wq; bias = bq; out = qb; }
  else if (blockIdx.z == 1) { W = Wk; bias = bk; out = kb; }
  else                      { W = Wv; bias = bv; out = vtb; }

  const int tid = threadIdx.x, wave = tid >> 5, lane = tid & 31;
  const int n0 = blockIdx.x * BN;
  const int m0 = blockIdx.y * BM;

  __shared__ unsigned short lds_a[BM * LDSK];
  __shared__ unsigned short lds_b[BN * LDSK];

  v8f zero = {};
  v8f c[4][2];
#pragma unroll
  for (int i = 0; i < 4; ++i)
#pragma unroll
    for (int j = 0; j < 2; ++j) c[i][j] = zero;

  for (int k0 = 0; k0 < DD; k0 += BK) {
    if (k0 + BK < DD) {
      __builtin_prefetch(x + (size_t)m0 * DD + (k0 + BK), 0, 1);
      __builtin_prefetch(W + (size_t)(k0 + BK) * HH + n0, 0, 1);
    }
    ld_f32_rows(lds_a, x + (size_t)m0 * DD + k0, DD, tid);
    ld_f32_trans(lds_b, W + (size_t)k0 * HH + n0, HH, tid);
    __syncthreads();
    mma_step(lds_a, lds_b, c, wave, lane);
    __syncthreads();
  }

  const int wm = (wave >> 2) * 64, wn = (wave & 3) * 32;
  const int ln = lane & 15, lh = lane >> 4;
  const bool trans = (blockIdx.z == 2);
#pragma unroll
  for (int j = 0; j < 2; ++j) {
    const int col = n0 + wn + j * 16 + ln;     // h
    const float bj = bias[col];
#pragma unroll
    for (int i = 0; i < 4; ++i)
#pragma unroll
      for (int vi = 0; vi < 8; ++vi) {
        const int row = m0 + wm + i * 16 + vi + lh * 8;   // b*S + t
        const unsigned short val = f2bf(c[i][j][vi] + bj);
        if (trans) {
          // v stored as [b][h][t] so PV's B-tile is a plain row-major copy
          const int bz = row >> 11;            // / SS
          const int t  = row & (SS - 1);
          out[((size_t)bz * HH + col) * SS + t] = val;
        } else {
          out[(size_t)row * HH + col] = val;
        }
      }
  }
}

// ============ kernel 2: scores = (q @ k^T) / 32, f32 out (async double-buffered) ============
__global__ __launch_bounds__(256) void attn_qk_kernel(
    const unsigned short* __restrict__ qb, const unsigned short* __restrict__ kb,
    float* __restrict__ scores) {
  const int tid = threadIdx.x, wave = tid >> 5, lane = tid & 31;
  const int n0 = blockIdx.x * BN;            // t
  const int m0 = blockIdx.y * BM;            // s
  const int bz = blockIdx.z;

  const unsigned short* A = qb + ((size_t)bz * SS + m0) * HH;   // q rows
  const unsigned short* B = kb + ((size_t)bz * SS + n0) * HH;   // k rows == k^T cols

  __shared__ unsigned short lds_a[2][BM * LDSK];
  __shared__ unsigned short lds_b[2][BN * LDSK];

  v8f zero = {};
  v8f c[4][2];
#pragma unroll
  for (int i = 0; i < 4; ++i)
#pragma unroll
    for (int j = 0; j < 2; ++j) c[i][j] = zero;

  gemm_core(A, HH, B, HH, HH, lds_a, lds_b, c, tid, wave, lane);

  const int wm = (wave >> 2) * 64, wn = (wave & 3) * 32;
  const int ln = lane & 15, lh = lane >> 4;
  const float scale = 0.03125f;              // 1/sqrt(1024)
#pragma unroll
  for (int i = 0; i < 4; ++i)
#pragma unroll
    for (int j = 0; j < 2; ++j)
#pragma unroll
      for (int vi = 0; vi < 8; ++vi) {
        const int row = m0 + wm + i * 16 + vi + lh * 8;
        const int col = n0 + wn + j * 16 + ln;
        scores[((size_t)bz * SS + row) * SS + col] = c[i][j][vi] * scale;
      }
}

// ============ kernel 3: row softmax f32 -> bf16 probs ============
__global__ __launch_bounds__(256) void attn_softmax_kernel(
    const float* __restrict__ scores, unsigned short* __restrict__ probs) {
  const int tid = threadIdx.x;
  const size_t row = blockIdx.x;             // 0 .. B*S-1
  const float* src = scores + row * SS;

  float vals[8];
#pragma unroll
  for (int i = 0; i < 8; ++i) vals[i] = src[tid + i * 256];

  __shared__ float red[256];
  float m = -3.402823466e38f;
#pragma unroll
  for (int i = 0; i < 8; ++i) m = fmaxf(m, vals[i]);
  red[tid] = m; __syncthreads();
  for (int off = 128; off > 0; off >>= 1) {
    if (tid < off) red[tid] = fmaxf(red[tid], red[tid + off]);
    __syncthreads();
  }
  m = red[0]; __syncthreads();

  float s = 0.f;
#pragma unroll
  for (int i = 0; i < 8; ++i) { vals[i] = __expf(vals[i] - m); s += vals[i]; }
  red[tid] = s; __syncthreads();
  for (int off = 128; off > 0; off >>= 1) {
    if (tid < off) red[tid] += red[tid + off];
    __syncthreads();
  }
  const float inv = 1.0f / red[0];

  unsigned short* dst = probs + row * SS;
#pragma unroll
  for (int i = 0; i < 8; ++i) dst[tid + i * 256] = f2bf(vals[i] * inv);
}

// ============ kernel 4: out = probs @ v, f32 out (async double-buffered, vT rows) ============
__global__ __launch_bounds__(256) void attn_pv_kernel(
    const unsigned short* __restrict__ probs, const unsigned short* __restrict__ vtb,
    float* __restrict__ out) {
  const int tid = threadIdx.x, wave = tid >> 5, lane = tid & 31;
  const int n0 = blockIdx.x * BN;            // h
  const int m0 = blockIdx.y * BM;            // s
  const int bz = blockIdx.z;

  const unsigned short* A = probs + ((size_t)bz * SS + m0) * SS;        // probs rows [s][t]
  const unsigned short* B = vtb + ((size_t)bz * HH + n0) * SS;          // vT rows [h][t]

  __shared__ unsigned short lds_a[2][BM * LDSK];
  __shared__ unsigned short lds_b[2][BN * LDSK];

  v8f zero = {};
  v8f c[4][2];
#pragma unroll
  for (int i = 0; i < 4; ++i)
#pragma unroll
    for (int j = 0; j < 2; ++j) c[i][j] = zero;

  gemm_core(A, SS, B, SS, SS, lds_a, lds_b, c, tid, wave, lane);

  const int wm = (wave >> 2) * 64, wn = (wave & 3) * 32;
  const int ln = lane & 15, lh = lane >> 4;
#pragma unroll
  for (int i = 0; i < 4; ++i)
#pragma unroll
    for (int j = 0; j < 2; ++j)
#pragma unroll
      for (int vi = 0; vi < 8; ++vi) {
        const int row = m0 + wm + i * 16 + vi + lh * 8;
        const int col = n0 + wn + j * 16 + ln;
        out[((size_t)bz * SS + row) * HH + col] = c[i][j][vi];
      }
}

// ---------------- launch ----------------
extern "C" void kernel_launch(void* const* d_in, const int* in_sizes, int n_in,
                              void* d_out, int out_size, void* d_ws, size_t ws_size,
                              hipStream_t stream) {
  (void)in_sizes; (void)n_in; (void)out_size; (void)ws_size;
  const float* x  = (const float*)d_in[0];
  const float* Wq = (const float*)d_in[1];
  const float* bq = (const float*)d_in[2];
  const float* Wk = (const float*)d_in[3];
  const float* bk = (const float*)d_in[4];
  const float* Wv = (const float*)d_in[5];
  const float* bv = (const float*)d_in[6];

  // workspace layout (bytes): q 16M | k 16M | vT 16M | probs 32M | scores 64M  = 144MB
  unsigned short* qb    = (unsigned short*)d_ws;
  unsigned short* kb    = qb + (size_t)NROWS * HH;
  unsigned short* vtb   = kb + (size_t)NROWS * HH;
  unsigned short* probs = vtb + (size_t)NROWS * HH;
  float*          scores = (float*)(probs + (size_t)BB * SS * SS);

  dim3 blk(256);
  attn_qkv_kernel<<<dim3(HH / BN, NROWS / BM, 3), blk, 0, stream>>>(
      x, Wq, bq, Wk, bk, Wv, bv, qb, kb, vtb);
  attn_qk_kernel<<<dim3(SS / BN, SS / BM, BB), blk, 0, stream>>>(qb, kb, scores);
  attn_softmax_kernel<<<dim3(NROWS), blk, 0, stream>>>(scores, probs);
  attn_pv_kernel<<<dim3(HH / BN, SS / BM, BB), blk, 0, stream>>>(probs, vtb, (float*)d_out);
}
